// DeepCache_73512660238874
// MI455X (gfx1250) — matrix operand order repaired and checked by reading, not compiled, not run on hardware
//
#include <hip/hip_runtime.h>
#include <hip/hip_bf16.h>

// ---------------- CDNA5 WMMA types ----------------
typedef _Float16 v16h __attribute__((ext_vector_type(16)));
typedef _Float16 v8h  __attribute__((ext_vector_type(8)));
typedef float    v8f  __attribute__((ext_vector_type(8)));

#define B_   64
#define S_   4096
#define EMB_ 20
#define HID_ 20
#define VOC_ 256

// packed-weight tile ids: enc (3kc x 2nt) -> 0..5, W_ih (2kc x 5nt) -> 6..15,
// W_hh (5nt) -> 16..20.  Each tile: 32 lanes x 16 f16 = 512 halfs = 1KB.
#define TILE_ENC(kc, nt)  ((kc) * 2 + (nt))
#define TILE_WIH(kc, nt)  (6 + (kc) * 5 + (nt))
#define TILE_WHH(nt)      (16 + (nt))

// hardware reciprocal: single v_rcp_f32 instead of the IEEE-div expansion
static __device__ __forceinline__ float fast_rcp(float x) {
    return __builtin_amdgcn_rcpf(x);
}
static __device__ __forceinline__ float sigm(float x) {
    return fast_rcp(1.0f + __expf(-x));
}
static __device__ __forceinline__ float fast_tanh(float x) {
    x = fminf(fmaxf(x, -15.0f), 15.0f);     // keep exp finite
    float e = __expf(-2.0f * x);
    return (1.0f - e) * fast_rcp(1.0f + e);
}

// --- A-matrix (16x32 f16) from row-major LDS tile: two contiguous 16B runs ---
// ISA 7.12.2: lane m=l&15, K-half kh=l>>4; elems 0..7 = K kh*8..+7,
// elems 8..15 = K 16+kh*8..+7.
static __device__ __forceinline__ v16h load_A16(const _Float16* x, int stride,
                                                int l, int kc) {
    int m  = l & 15;
    int kh = l >> 4;
    const _Float16* row = x + m * stride + kc * 32;
    v8h lo = *(const v8h*)(row + kh * 8);
    v8h hi = *(const v8h*)(row + 16 + kh * 8);
    v16h a;
#pragma unroll
    for (int i = 0; i < 8; ++i) {
        a[i]     = lo[i];
        a[i + 8] = hi[i];
    }
    return a;
}

// --- generic (slow, branchy) B-tile builder: used ONLY in the pack kernel ---
// B[k][n] = W[n][k] (x @ W^T). elems 0..7 = K kh*8..+7, elems 8..15 = K 16+kh*8..+7,
// lane%16 = N. Zero-pad outside kmax/nmax.
static __device__ __forceinline__ v16h build_B16(const float* __restrict__ W,
                                                 int wcols, int kmax, int nmax,
                                                 int l, int kc, int nt) {
    int nl = l & 15;
    int kg = l >> 4;
    int n  = nt * 16 + nl;
    v16h bm;
#pragma unroll
    for (int v = 0; v < 8; ++v) {
        int k = kc * 32 + (v >> 2) * 16 + kg * 8 + 2 * (v & 3);
        float b0 = (n < nmax && k     < kmax) ? W[(size_t)n * wcols + k]     : 0.f;
        float b1 = (n < nmax && k + 1 < kmax) ? W[(size_t)n * wcols + k + 1] : 0.f;
        bm[2 * v]     = (_Float16)b0;
        bm[2 * v + 1] = (_Float16)b1;
    }
    return bm;
}

// --- fast path: one 32-byte vector load per B operand, no branches ---
static __device__ __forceinline__ v16h load_Btile(
    const _Float16* __restrict__ wp, int tile, int l) {
    return ((const v16h*)(wp + (size_t)tile * 512))[l];
}

// pack kernel: run once per launch, 1 wave
__global__ __launch_bounds__(32) void k_pack(const float* __restrict__ enc_W,
                                             const float* __restrict__ W_ih,
                                             const float* __restrict__ W_hh,
                                             _Float16* __restrict__ wp) {
    int l = threadIdx.x;
    for (int kc = 0; kc < 3; ++kc)
        for (int nt = 0; nt < 2; ++nt)
            ((v16h*)(wp + TILE_ENC(kc, nt) * 512))[l] =
                build_B16(enc_W, 80, 80, 20, l, kc, nt);
    for (int kc = 0; kc < 2; ++kc)
        for (int nt = 0; nt < 5; ++nt)
            ((v16h*)(wp + TILE_WIH(kc, nt) * 512))[l] =
                build_B16(W_ih, 40, 40, 80, l, kc, nt);
    for (int nt = 0; nt < 5; ++nt)
        ((v16h*)(wp + TILE_WHH(nt) * 512))[l] =
            build_B16(W_hh, 20, 20, 80, l, 0, nt);
}

// epilogue for one encoder: sigmoid(acc+bias) -> embf LDS, and dist partial sums
static __device__ __forceinline__ void enc_epilogue(v8f a0, v8f a1, int colbase,
                                                    int l, int b,
                                                    const float* __restrict__ enc_b,
                                                    float* __restrict__ dsum,
                                                    _Float16 (*embf)[64]) {
    int kh = l >> 4;
#pragma unroll
    for (int nt = 0; nt < 2; ++nt) {
        v8f a = nt ? a1 : a0;
        int N = nt * 16 + (l & 15);
        bool valid = (N < 20);
        float bias = valid ? enc_b[N] : 0.f;
        float rs = 0.f;
#pragma unroll
        for (int r = 0; r < 8; ++r) {
            float vs = sigm(a[r] + bias);
            if (valid) {
                embf[r + 8 * kh][colbase + N] = (_Float16)vs;
                rs += vs;
            }
        }
        rs += __shfl_xor(rs, 16);           // combine the two 8-row halves
        if (valid && kh == 0)
            atomicAdd(&dsum[b * 40 + colbase + N], rs);
    }
}

union HPack {
    _Float16 h[2];
    unsigned u;
};

// ===========================================================================
// Kernel 1: byte-embed + 2 encoders + LSTM input projection (pre-gates).
// One wave per 16-row (b, t0..t0+15) tile. 22 WMMAs per workgroup.
// Pre-gates are scattered into the LSTM-consumption layout:
//   P[b>>4][t][lane][slot], lane = (n&15) + 16*((b>>3)&1), slot = (n>>4)*8+(b&7)
// ===========================================================================
__global__ __launch_bounds__(32) void k_embed_encode_pregate(
    const int* __restrict__ inp,
    const float* __restrict__ pc_emb, const float* __restrict__ addr_emb,
    const _Float16* __restrict__ wp,  const float* __restrict__ enc_b,
    const float* __restrict__ b_ih,   const float* __restrict__ b_hh,
    _Float16* __restrict__ pregate,   float* __restrict__ dsum) {
    __shared__ _Float16 xpc[16][96];   // 16x80 gathered embeddings, K-pad to 96
    __shared__ _Float16 xad[16][96];
    __shared__ _Float16 embf[16][64];  // 16x40 sigmoid outputs, K-pad to 64

    const int l  = threadIdx.x;
    const int b  = blockIdx.y;
    const int t0 = blockIdx.x * 16;

    // ---- gather: each lane fills half a row (two 20-wide i-groups) ----
    {
        int r = l >> 1, half = l & 1;
        unsigned vpc = (unsigned)inp[((size_t)b * S_ + t0 + r) * 2 + 0];
        unsigned vad = (unsigned)inp[((size_t)b * S_ + t0 + r) * 2 + 1];
#pragma unroll
        for (int g = 0; g < 2; ++g) {
            int i    = half * 2 + g;
            int col0 = i * 20;
            unsigned bp = (vpc >> (24 - 8 * i)) & 0xFFu;
            unsigned ba = (vad >> (24 - 8 * i)) & 0xFFu;
            const float2* sp = (const float2*)&pc_emb[((size_t)i * VOC_ + bp) * EMB_];
            const float2* sa = (const float2*)&addr_emb[((size_t)i * VOC_ + ba) * EMB_];
#pragma unroll
            for (int e2 = 0; e2 < 10; ++e2) {
                float2 fp = sp[e2], fa = sa[e2];
                HPack hp, ha;
                hp.h[0] = (_Float16)fp.x; hp.h[1] = (_Float16)fp.y;
                ha.h[0] = (_Float16)fa.x; ha.h[1] = (_Float16)fa.y;
                *(unsigned*)&xpc[r][col0 + 2 * e2] = hp.u;
                *(unsigned*)&xad[r][col0 + 2 * e2] = ha.u;
            }
        }
        // zero pads (all even columns -> u32 stores)
#pragma unroll
        for (int c = 0; c < 8; c += 2)
            *(unsigned*)&xpc[r][80 + half * 8 + c] = 0u;
#pragma unroll
        for (int c = 0; c < 8; c += 2)
            *(unsigned*)&xad[r][80 + half * 8 + c] = 0u;
#pragma unroll
        for (int c = 0; c < 12; c += 2)
            *(unsigned*)&embf[r][40 + half * 12 + c] = 0u;
    }
    __syncthreads();

    // ---- encoders: (16x80) @ (80x20), K in 3 chunks, N in 2 tiles ----
    v8f accP[2] = {}, accA[2] = {};
#pragma unroll
    for (int kc = 0; kc < 3; ++kc) {
        v16h aP = load_A16(&xpc[0][0], 96, l, kc);
        v16h aA = load_A16(&xad[0][0], 96, l, kc);
#pragma unroll
        for (int nt = 0; nt < 2; ++nt) {
            v16h bm = load_Btile(wp, TILE_ENC(kc, nt), l);
            accP[nt] = __builtin_amdgcn_wmma_f32_16x16x32_f16(
                false, aP, false, bm, (short)0, accP[nt], false, false);
            accA[nt] = __builtin_amdgcn_wmma_f32_16x16x32_f16(
                false, aA, false, bm, (short)0, accA[nt], false, false);
        }
    }
    enc_epilogue(accP[0], accP[1], 0,  l, b, enc_b, dsum, embf);
    enc_epilogue(accA[0], accA[1], 20, l, b, enc_b, dsum, embf);
    __syncthreads();

    // ---- pre-gates: (16x40) @ (40x80), K in 2 chunks, N in 5 tiles ----
    v8f pacc[5] = {};
#pragma unroll
    for (int kc = 0; kc < 2; ++kc) {
        v16h aE = load_A16(&embf[0][0], 64, l, kc);
#pragma unroll
        for (int nt = 0; nt < 5; ++nt) {
            v16h bm = load_Btile(wp, TILE_WIH(kc, nt), l);
            pacc[nt] = __builtin_amdgcn_wmma_f32_16x16x32_f16(
                false, aE, false, bm, (short)0, pacc[nt], false, false);
        }
    }
    // scatter into LSTM-consumption layout
    const int kh    = l >> 4;
    const int bt    = b >> 4;
    const int lane2 = (l & 15) + 16 * ((b >> 3) & 1);
    const int bslot = b & 7;
#pragma unroll
    for (int nt = 0; nt < 5; ++nt) {
        int n = nt * 16 + (l & 15);
        float bias = b_ih[n] + b_hh[n];
#pragma unroll
        for (int r = 0; r < 8; ++r) {
            int t = t0 + r + 8 * kh;
            pregate[(((size_t)bt * S_ + t) * 32 + lane2) * 40 + nt * 8 + bslot] =
                (_Float16)(pacc[nt][r] + bias);
        }
    }
}

// ===========================================================================
// Kernel 2: LSTM scan. 4 persistent waves, one 16-row batch tile each.
// Per step: 5 contiguous 16B pre-gate loads -> C, 5 WMMAs for h @ W_hh^T,
// LDS shuffle, per-lane (row,hid) gate math with f32 (h,c) state.
// ===========================================================================
__global__ __launch_bounds__(32) void k_lstm(
    const _Float16* __restrict__ pregate, const _Float16* __restrict__ wp,
    const float* __restrict__ h0, const float* __restrict__ c0,
    float* __restrict__ hfinal) {
    __shared__ float    gbuf[16][80];
    __shared__ _Float16 hbuf[16][32];   // K-padded h tile for the A operand

    const int l  = threadIdx.x;
    const int b0 = blockIdx.x * 16;
    const int nl = l & 15, kh = l >> 4;

    // recurrent weights as 5 resident B tiles (K=20 padded to 32)
    v16h Bh[5];
#pragma unroll
    for (int nt = 0; nt < 5; ++nt) Bh[nt] = load_Btile(wp, TILE_WHH(nt), l);

    for (int i = l; i < 16 * 32; i += 32) (&hbuf[0][0])[i] = (_Float16)0.f;

    int   rj[10], hj[10];
    float creg[10], hreg[10];
#pragma unroll
    for (int j = 0; j < 10; ++j) {          // 320 (row,hid) pairs / 32 lanes
        int p = l + 32 * j;
        rj[j] = p / 20;
        hj[j] = p % 20;
        creg[j] = c0[(b0 + rj[j]) * HID_ + hj[j]];
        hreg[j] = h0[(b0 + rj[j]) * HID_ + hj[j]];
    }
    __syncthreads();
#pragma unroll
    for (int j = 0; j < 10; ++j) hbuf[rj[j]][hj[j]] = (_Float16)hreg[j];
    __syncthreads();

    // per-lane pre-gate cursor: 80 contiguous bytes per lane per step
    const _Float16* pl = pregate + (size_t)blockIdx.x * S_ * 1280 + l * 40;

    for (int t = 0; t < S_; ++t) {
        __builtin_prefetch(pl + 8 * 1280, 0, 1);   // stream 8 steps ahead

        // C <- pre-gates (f16 -> f32), already in WMMA C layout order
        v8h pg[5];
#pragma unroll
        for (int nt = 0; nt < 5; ++nt) pg[nt] = *(const v8h*)(pl + nt * 8);

        v8f acc[5];
#pragma unroll
        for (int nt = 0; nt < 5; ++nt)
#pragma unroll
            for (int r = 0; r < 8; ++r) acc[nt][r] = (float)pg[nt][r];

        v16h hA = load_A16(&hbuf[0][0], 32, l, 0);
#pragma unroll
        for (int nt = 0; nt < 5; ++nt)
            acc[nt] = __builtin_amdgcn_wmma_f32_16x16x32_f16(
                false, hA, false, Bh[nt], (short)0, acc[nt], false, false);

#pragma unroll
        for (int nt = 0; nt < 5; ++nt)
#pragma unroll
            for (int r = 0; r < 8; ++r)
                gbuf[r + 8 * kh][nt * 16 + nl] = acc[nt][r];
        __syncthreads();

#pragma unroll
        for (int j = 0; j < 10; ++j) {      // torch gate order i,f,g,o
            float gi = gbuf[rj[j]][hj[j]];
            float gf = gbuf[rj[j]][20 + hj[j]];
            float gg = gbuf[rj[j]][40 + hj[j]];
            float go = gbuf[rj[j]][60 + hj[j]];
            float c  = sigm(gf) * creg[j] + sigm(gi) * fast_tanh(gg);
            float h  = sigm(go) * fast_tanh(c);
            creg[j]  = c;
            hreg[j]  = h;
            hbuf[rj[j]][hj[j]] = (_Float16)h;
        }
        __syncthreads();
        pl += 1280;
    }
#pragma unroll
    for (int j = 0; j < 10; ++j)
        hfinal[(b0 + rj[j]) * HID_ + hj[j]] = hreg[j];
}

// ===========================================================================
// Kernel 3a: logits + softmax over the BATCH axis (T = 1e-3, max-subtracted).
// ===========================================================================
__global__ __launch_bounds__(32) void k_logits_softmax(
    const float* __restrict__ hfinal, const float* __restrict__ dec_W,
    const float* __restrict__ dec_b, float* __restrict__ out_probs,
    float* __restrict__ out_logits) {
    const int l = threadIdx.x;
    const int k = blockIdx.x >> 8;
    const int v = blockIdx.x & 255;
    const float* w = dec_W + ((size_t)k * VOC_ + v) * HID_;
    const float bias = dec_b[k * VOC_ + v];

    float lg[2];
#pragma unroll
    for (int i = 0; i < 2; ++i) {
        int bb = l + 32 * i;
        const float* h = hfinal + bb * HID_;
        float s = bias;
#pragma unroll
        for (int j = 0; j < HID_; ++j) s += h[j] * w[j];
        lg[i] = s;
        out_logits[((size_t)k * B_ + bb) * VOC_ + v] = s;
    }
    float m = fmaxf(lg[0], lg[1]);
    for (int off = 16; off > 0; off >>= 1) m = fmaxf(m, __shfl_xor(m, off));
    float e0 = __expf((lg[0] - m) * 1000.0f);
    float e1 = __expf((lg[1] - m) * 1000.0f);
    float s = e0 + e1;
    for (int off = 16; off > 0; off >>= 1) s += __shfl_xor(s, off);
    float rs = fast_rcp(s);
    out_probs[((size_t)k * B_ + l) * VOC_ + v]      = e0 * rs;
    out_probs[((size_t)k * B_ + l + 32) * VOC_ + v] = e1 * rs;
}

// Kernel 3b: byte_e[b][k*20+e] = sum_v probs[k,b,v] * addr_emb[k,v,e]
__global__ void k_byte_e(const float* __restrict__ probs,
                         const float* __restrict__ addr_emb,
                         float* __restrict__ byte_e) {
    int idx = blockIdx.x * blockDim.x + threadIdx.x;  // 64*4*20 = 5120
    if (idx >= B_ * 4 * EMB_) return;
    int e = idx % 20, kk = (idx / 20) & 3, b = idx / 80;
    float s = 0.f;
    for (int v = 0; v < VOC_; ++v)
        s += probs[((size_t)kk * B_ + b) * VOC_ + v] *
             addr_emb[((size_t)kk * VOC_ + v) * EMB_ + e];
    byte_e[b * 80 + kk * 20 + e] = s;
}

// Kernel 3c: fe = sigmoid(byte_e @ enc_W^T + enc_b); MLP head -> freq_rec
__global__ __launch_bounds__(64) void k_head(
    const float* __restrict__ byte_e, const float* __restrict__ dsum,
    const float* __restrict__ enc_W, const float* __restrict__ enc_b,
    const float* __restrict__ d1_W, const float* __restrict__ d1_b,
    const float* __restrict__ d2_W, const float* __restrict__ d2_b,
    float* __restrict__ out_f0, float* __restrict__ out_f1) {
    int b = blockIdx.x * blockDim.x + threadIdx.x;
    if (b >= B_) return;
    float fr[60];
    for (int n = 0; n < 20; ++n) {
        float s = enc_b[n];
        for (int kx = 0; kx < 80; ++kx) s += byte_e[b * 80 + kx] * enc_W[n * 80 + kx];
        fr[n] = sigm(s);
    }
    for (int d = 0; d < 40; ++d) fr[20 + d] = dsum[b * 40 + d] * (1.0f / (float)S_);
    float h1[10];
    for (int m = 0; m < 10; ++m) {
        float s = d1_b[m];
        for (int j = 0; j < 60; ++j) s += fr[j] * d1_W[m * 60 + j];
        h1[m] = fmaxf(s, 0.f);
    }
    float o0 = d2_b[0], o1 = d2_b[1];
    for (int m = 0; m < 10; ++m) {
        o0 += h1[m] * d2_W[m];
        o1 += h1[m] * d2_W[10 + m];
    }
    out_f0[b] = o0;
    out_f1[b] = o1;
}

__global__ void k_init(float* __restrict__ dsum, float* __restrict__ byte_e) {
    int i = blockIdx.x * blockDim.x + threadIdx.x;
    if (i < B_ * 40) dsum[i] = 0.f;
    if (i < B_ * 80) byte_e[i] = 0.f;
}

// ===========================================================================
extern "C" void kernel_launch(void* const* d_in, const int* in_sizes, int n_in,
                              void* d_out, int out_size, void* d_ws,
                              size_t ws_size, hipStream_t stream) {
    const int*   inp      = (const int*)d_in[0];
    const float* h0       = (const float*)d_in[1];
    const float* c0       = (const float*)d_in[2];
    const float* pc_emb   = (const float*)d_in[3];
    const float* addr_emb = (const float*)d_in[4];
    const float* enc_W    = (const float*)d_in[5];
    const float* enc_b    = (const float*)d_in[6];
    const float* W_ih     = (const float*)d_in[7];
    const float* W_hh     = (const float*)d_in[8];
    const float* b_ih     = (const float*)d_in[9];
    const float* b_hh     = (const float*)d_in[10];
    const float* dec_W    = (const float*)d_in[11];
    const float* dec_b    = (const float*)d_in[12];
    const float* d1_W     = (const float*)d_in[13];
    const float* d1_b     = (const float*)d_in[14];
    const float* d2_W     = (const float*)d_in[15];
    const float* d2_b     = (const float*)d_in[16];

    char* ws = (char*)d_ws;
    // pregate f16 (packed LSTM layout): B*S*80 = 20,971,520 elems = 41,943,040 B
    _Float16* pregate = (_Float16*)ws;
    float*    dsum    = (float*)(ws + 41943040);          // 64*40 f32
    float*    hfinal  = (float*)(ws + 41953280);          // 64*20 f32
    float*    byte_e  = (float*)(ws + 41958400);          // 64*80 f32
    _Float16* wpack   = (_Float16*)(ws + 41978880);       // 21 tiles * 1KB

    float* out        = (float*)d_out;
    float* out_probs  = out;                // (4,1,64,256) = 65536
    float* out_logits = out + 65536;        // (4,1,64,256) = 65536
    float* out_f0     = out + 131072;       // (64,)
    float* out_f1     = out + 131136;       // (64,)

    k_init<<<20, 256, 0, stream>>>(dsum, byte_e);
    k_pack<<<1, 32, 0, stream>>>(enc_W, W_ih, W_hh, wpack);

    dim3 g1(S_ / 16, B_);                   // 256 x 64 tiles, 1 wave each
    k_embed_encode_pregate<<<g1, 32, 0, stream>>>(
        inp, pc_emb, addr_emb, wpack, enc_b, b_ih, b_hh, pregate, dsum);

    k_lstm<<<4, 32, 0, stream>>>(pregate, wpack, h0, c0, hfinal);

    k_logits_softmax<<<4 * VOC_, 32, 0, stream>>>(hfinal, dec_W, dec_b,
                                                  out_probs, out_logits);
    k_byte_e<<<20, 256, 0, stream>>>(out_probs, addr_emb, byte_e);
    k_head<<<1, 64, 0, stream>>>(byte_e, dsum, enc_W, enc_b, d1_W, d1_b, d2_W,
                                 d2_b, out_f0, out_f1);
}
